// DHHT_47107201303148
// MI455X (gfx1250) — matrix operand order repaired and checked by reading, not compiled, not run on hardware
//
#include <hip/hip_runtime.h>
#include <cstddef>

#define NTOK 32768
#define DINP 257
#define DOUT 256
#define NH   8
#define HD   2048    // NH*DOUT
#define EPSF 1e-6f
#define KCURV 1.0f
#define SPLITK 16
#define CHUNK (NTOK / SPLITK)   // 2048
#define KSPLIT 64

typedef __attribute__((ext_vector_type(16))) __bf16 v16bf;
typedef __attribute__((ext_vector_type(8)))  __bf16 v8bf;
typedef __attribute__((ext_vector_type(4)))  __bf16 v4bf;
typedef __attribute__((ext_vector_type(8)))  float  v8f;
typedef __attribute__((ext_vector_type(4)))  float  v4f;

// ---------------------------------------------------------------------------
// WMMA fragment builders (ISA 7.12.2 layouts, wave32), vectorized LDS reads.
// A 16x32 bf16 from a row-major tile: two contiguous v8bf (16B) loads/lane.
// ---------------------------------------------------------------------------
static __device__ inline v16bf load_frag_a(const __bf16* A, int lda) {
  const int lane = threadIdx.x & 31;
  const int m = lane & 15;
  const int khalf = (lane >> 4) * 8;
  const __bf16* row = A + m * lda;
  v8bf lo = *(const v8bf*)(row + khalf);
  v8bf hi = *(const v8bf*)(row + 16 + khalf);
  return __builtin_shufflevector(lo, hi, 0, 1, 2, 3, 4, 5, 6, 7,
                                 8, 9, 10, 11, 12, 13, 14, 15);
}

// B 32x16 bf16 from a column-major tile (B[n*32+k]): two v8bf (16B) loads/lane.
static __device__ inline v16bf load_frag_b32(const __bf16* B) {
  const int lane = threadIdx.x & 31;
  const int n = lane & 15;
  const int koff = (lane >> 4) * 8;
  v8bf lo = *(const v8bf*)(B + n * 32 + koff);
  v8bf hi = *(const v8bf*)(B + n * 32 + 16 + koff);
  return __builtin_shufflevector(lo, hi, 0, 1, 2, 3, 4, 5, 6, 7,
                                 8, 9, 10, 11, 12, 13, 14, 15);
}

#define WMMA_BF16(a, b, c) \
  __builtin_amdgcn_wmma_f32_16x16x32_bf16(false, (a), false, (b), (short)0, (c), false, false)

// ---------------------------------------------------------------------------
// K0: phi_q = fp2((relu(query @ Wq_h^T + bq)+eps)/dscale), bf16 out
// grid (NTOK/32, NH), block 256 (8 waves). Block tile: 32 rows x 256 cols.
// ---------------------------------------------------------------------------
__global__ __launch_bounds__(256)
void proj_phi_q_kernel(const float* __restrict__ x, const float* __restrict__ W,
                       const float* __restrict__ bias, const float* __restrict__ nscale,
                       __bf16* __restrict__ phi_out) {
  const int n0 = blockIdx.x * 32;
  const int h  = blockIdx.y;
  const int tid = threadIdx.x;
  const int wave = tid >> 5, lane = tid & 31;
  const int rowtile = wave & 1;
  const int colbase = (wave >> 1) * 64;
  const int crow0 = (lane >> 4) * 8, ccol = lane & 15;

  __shared__ __attribute__((aligned(32))) __bf16 As[32 * 32];    // [row][k]
  __shared__ __attribute__((aligned(32))) __bf16 Bs[256 * 32];   // [o][k] col-major
  __shared__ float s2[32], s4[32];

  v8f acc[4];
#pragma unroll
  for (int t = 0; t < 4; ++t)
#pragma unroll
    for (int i = 0; i < 8; ++i) acc[t][i] = 0.0f;

  for (int kt = 0; kt < 9; ++kt) {            // K = 257 padded to 288
    const int k0 = kt * 32;
    __syncthreads();
    if (k0 + 32 <= DINP) {                    // full tiles: branch-free staging
      for (int i = tid; i < 32 * 32; i += 256) {
        int k = i & 31, r = i >> 5;
        As[r * 32 + k] = (__bf16)x[(size_t)(n0 + r) * DINP + k0 + k];
      }
      for (int i = tid; i < 256 * 32; i += 256) {
        int kk = i & 31, o = i >> 5;
        Bs[o * 32 + kk] = (__bf16)W[((size_t)h * DOUT + o) * DINP + k0 + kk];
      }
    } else {                                  // tail tile (k = 256 only)
      for (int i = tid; i < 32 * 32; i += 256) {
        int k = i & 31, r = i >> 5;
        int kidx = k0 + k;
        float v = x[(size_t)(n0 + r) * DINP + (kidx < DINP ? kidx : DINP - 1)];
        As[r * 32 + k] = (__bf16)(kidx < DINP ? v : 0.0f);
      }
      for (int i = tid; i < 256 * 32; i += 256) {
        int kk = i & 31, o = i >> 5;
        int kidx = k0 + kk;
        float v = W[((size_t)h * DOUT + o) * DINP + (kidx < DINP ? kidx : DINP - 1)];
        Bs[o * 32 + kk] = (__bf16)(kidx < DINP ? v : 0.0f);
      }
    }
    __syncthreads();
    v16bf a = load_frag_a(As + rowtile * 16 * 32, 32);
#pragma unroll
    for (int t = 0; t < 4; ++t) {
      v16bf b = load_frag_b32(Bs + (colbase + t * 16) * 32);
      acc[t] = WMMA_BF16(a, b, acc[t]);
    }
  }
  __syncthreads();
  if (tid < 32) { s2[tid] = 0.0f; s4[tid] = 0.0f; }
  __syncthreads();

  const float inv_ds = 1.0f / (fabsf(nscale[0]) + EPSF);
  float bcol[4];
#pragma unroll
  for (int t = 0; t < 4; ++t) bcol[t] = bias[h * DOUT + colbase + t * 16 + ccol];

#pragma unroll
  for (int r = 0; r < 8; ++r) {
    int row = rowtile * 16 + crow0 + r;
    float a2 = 0.0f, a4 = 0.0f;
#pragma unroll
    for (int t = 0; t < 4; ++t) {
      float xs = acc[t][r] + bcol[t];
      float tt = (fmaxf(xs, 0.0f) + EPSF) * inv_ds;
      float y = tt * tt;
      acc[t][r] = y;
      a2 += y;            // sum t^2
      a4 += y * y;        // sum t^4
    }
    atomicAdd(&s2[row], a2);
    atomicAdd(&s4[row], a4);
  }
  __syncthreads();
#pragma unroll
  for (int r = 0; r < 8; ++r) {
    int row = rowtile * 16 + crow0 + r;
    float sc = sqrtf(s2[row]) * rsqrtf(s4[row]);
#pragma unroll
    for (int t = 0; t < 4; ++t) {
      int col = colbase + t * 16 + ccol;
      phi_out[(size_t)(n0 + row) * HD + h * DOUT + col] = (__bf16)(sc * acc[t][r]);
    }
  }
}

// ---------------------------------------------------------------------------
// K1: K & V projections share the A tile; phi_k -> bf16, vs -> bf16
// ---------------------------------------------------------------------------
__global__ __launch_bounds__(256)
void proj_kv_kernel(const float* __restrict__ x,
                    const float* __restrict__ Wk, const float* __restrict__ bk,
                    const float* __restrict__ Wv, const float* __restrict__ bv,
                    const float* __restrict__ nscale,
                    __bf16* __restrict__ phi_k, __bf16* __restrict__ vs_bf) {
  const int n0 = blockIdx.x * 32;
  const int h  = blockIdx.y;
  const int tid = threadIdx.x;
  const int wave = tid >> 5, lane = tid & 31;
  const int rowtile = wave & 1;
  const int colbase = (wave >> 1) * 64;
  const int crow0 = (lane >> 4) * 8, ccol = lane & 15;

  __shared__ __attribute__((aligned(32))) __bf16 As[32 * 32];
  __shared__ __attribute__((aligned(32))) __bf16 Bk[256 * 32];
  __shared__ __attribute__((aligned(32))) __bf16 Bv[256 * 32];
  __shared__ float s2[32], s4[32];

  v8f accK[4], accV[4];
#pragma unroll
  for (int t = 0; t < 4; ++t)
#pragma unroll
    for (int i = 0; i < 8; ++i) { accK[t][i] = 0.0f; accV[t][i] = 0.0f; }

  for (int kt = 0; kt < 9; ++kt) {
    const int k0 = kt * 32;
    __syncthreads();
    if (k0 + 32 <= DINP) {
      for (int i = tid; i < 32 * 32; i += 256) {
        int k = i & 31, r = i >> 5;
        As[r * 32 + k] = (__bf16)x[(size_t)(n0 + r) * DINP + k0 + k];
      }
      for (int i = tid; i < 256 * 32; i += 256) {
        int kk = i & 31, o = i >> 5;
        size_t wi = ((size_t)h * DOUT + o) * DINP + k0 + kk;
        Bk[o * 32 + kk] = (__bf16)Wk[wi];
        Bv[o * 32 + kk] = (__bf16)Wv[wi];
      }
    } else {
      for (int i = tid; i < 32 * 32; i += 256) {
        int k = i & 31, r = i >> 5;
        int kidx = k0 + k;
        float v = x[(size_t)(n0 + r) * DINP + (kidx < DINP ? kidx : DINP - 1)];
        As[r * 32 + k] = (__bf16)(kidx < DINP ? v : 0.0f);
      }
      for (int i = tid; i < 256 * 32; i += 256) {
        int kk = i & 31, o = i >> 5;
        int kidx = k0 + kk;
        size_t wi = ((size_t)h * DOUT + o) * DINP + (kidx < DINP ? kidx : DINP - 1);
        bool ok = (kidx < DINP);
        float vk = Wk[wi], vv = Wv[wi];
        Bk[o * 32 + kk] = (__bf16)(ok ? vk : 0.0f);
        Bv[o * 32 + kk] = (__bf16)(ok ? vv : 0.0f);
      }
    }
    __syncthreads();
    v16bf a = load_frag_a(As + rowtile * 16 * 32, 32);
#pragma unroll
    for (int t = 0; t < 4; ++t) {
      v16bf b0 = load_frag_b32(Bk + (colbase + t * 16) * 32);
      accK[t] = WMMA_BF16(a, b0, accK[t]);
      v16bf b1 = load_frag_b32(Bv + (colbase + t * 16) * 32);
      accV[t] = WMMA_BF16(a, b1, accV[t]);
    }
  }
  __syncthreads();
  if (tid < 32) { s2[tid] = 0.0f; s4[tid] = 0.0f; }
  __syncthreads();

  const float inv_ds = 1.0f / (fabsf(nscale[0]) + EPSF);
  float bkc[4], bvc[4];
#pragma unroll
  for (int t = 0; t < 4; ++t) {
    bkc[t] = bk[h * DOUT + colbase + t * 16 + ccol];
    bvc[t] = bv[h * DOUT + colbase + t * 16 + ccol];
  }

#pragma unroll
  for (int r = 0; r < 8; ++r) {
    int row = rowtile * 16 + crow0 + r;
    float a2 = 0.0f, a4 = 0.0f;
#pragma unroll
    for (int t = 0; t < 4; ++t) {
      int col = colbase + t * 16 + ccol;
      float xk = accK[t][r] + bkc[t];
      float tt = (fmaxf(xk, 0.0f) + EPSF) * inv_ds;
      float y = tt * tt;
      accK[t][r] = y;
      a2 += y;
      a4 += y * y;
      vs_bf[(size_t)(n0 + row) * HD + h * DOUT + col] = (__bf16)(accV[t][r] + bvc[t]);
    }
    atomicAdd(&s2[row], a2);
    atomicAdd(&s4[row], a4);
  }
  __syncthreads();
#pragma unroll
  for (int r = 0; r < 8; ++r) {
    int row = rowtile * 16 + crow0 + r;
    float sc = sqrtf(s2[row]) * rsqrtf(s4[row]);
#pragma unroll
    for (int t = 0; t < 4; ++t) {
      int col = colbase + t * 16 + ccol;
      phi_k[(size_t)(n0 + row) * HD + h * DOUT + col] = (__bf16)(sc * accK[t][r]);
    }
  }
}

// ---------------------------------------------------------------------------
// K2: ktv[h] (256x256) = sum_n phi_k[n,h,:]^T vs[n,h,:], split-K + f32 atomics
// grid (NH, 4 output 128x128 tiles, SPLITK)
// ---------------------------------------------------------------------------
__global__ __launch_bounds__(256)
void ktv_kernel(const __bf16* __restrict__ phi_k, const __bf16* __restrict__ vs_bf,
                float* __restrict__ ktv) {
  const int h = blockIdx.x;
  const int mt = blockIdx.y >> 1, dt = blockIdx.y & 1;
  const int c0 = blockIdx.z * CHUNK;
  const int tid = threadIdx.x;
  const int wave = tid >> 5, lane = tid & 31;
  const int crow0 = (lane >> 4) * 8, ccol = lane & 15;

  __shared__ __attribute__((aligned(32))) __bf16 P[128 * 32];  // [m][k(n)] A tile
  __shared__ __attribute__((aligned(32))) __bf16 V[128 * 32];  // [d][k(n)] B tile

  v8f acc[8];
#pragma unroll
  for (int c = 0; c < 8; ++c)
#pragma unroll
    for (int i = 0; i < 8; ++i) acc[c][i] = 0.0f;

  for (int nt = 0; nt < CHUNK / 32; ++nt) {
    const int n0 = c0 + nt * 32;
    __syncthreads();
    // 16B-per-lane global loads (rows are 4KB-aligned), transpose into LDS
    for (int i = tid; i < (128 * 32) / 8; i += 256) {
      int m8 = (i & 15) * 8, kk = i >> 4;
      v8bf pv = *(const v8bf*)&phi_k[(size_t)(n0 + kk) * HD + h * DOUT + mt * 128 + m8];
      v8bf vv = *(const v8bf*)&vs_bf[(size_t)(n0 + kk) * HD + h * DOUT + dt * 128 + m8];
#pragma unroll
      for (int j = 0; j < 8; ++j) {
        P[(m8 + j) * 32 + kk] = pv[j];
        V[(m8 + j) * 32 + kk] = vv[j];
      }
    }
    __syncthreads();
    if (nt + 1 < CHUNK / 32) {  // prefetch next chunk -> global_prefetch_b8
      const int np = n0 + 32;
      __builtin_prefetch(&phi_k[(size_t)(np + (tid >> 3)) * HD + h * DOUT + mt * 128 + (tid & 7) * 16], 0, 1);
      __builtin_prefetch(&vs_bf[(size_t)(np + (tid >> 3)) * HD + h * DOUT + dt * 128 + (tid & 7) * 16], 0, 1);
    }
    v16bf a = load_frag_a(P + wave * 16 * 32, 32);
#pragma unroll
    for (int c = 0; c < 8; ++c) {
      v16bf b = load_frag_b32(V + (c * 16) * 32);
      acc[c] = WMMA_BF16(a, b, acc[c]);
    }
  }
#pragma unroll
  for (int c = 0; c < 8; ++c)
#pragma unroll
    for (int r = 0; r < 8; ++r) {
      int m = mt * 128 + wave * 16 + crow0 + r;
      int d = dt * 128 + c * 16 + ccol;
      unsafeAtomicAdd(&ktv[((size_t)h * DOUT + m) * DOUT + d], acc[c][r]);
    }
}

// K2b: ksum[h,m] = sum_n phi_k[n,h,m]
__global__ __launch_bounds__(256)
void ksum_kernel(const __bf16* __restrict__ phi_k, float* __restrict__ ksum) {
  const int h = blockIdx.x;
  const int c = blockIdx.y;
  const int m = threadIdx.x;
  const int rows = NTOK / KSPLIT;
  float s = 0.0f;
  for (int n = c * rows; n < (c + 1) * rows; ++n)
    s += (float)phi_k[(size_t)n * HD + h * DOUT + m];
  unsafeAtomicAdd(&ksum[h * DOUT + m], s);
}

// ---------------------------------------------------------------------------
// K3: per 32-row block:  for each head
//   num = phi_q @ ktv[h];  denom = phi_q . ksum[h];  attn = num/denom (+eps)
//   attn += vs @ v_map^T + v_map_b    (WMMA accumulates into divided num)
//   out  += attn @ final_w[:, h*256:(h+1)*256]^T
// epilogue: += final_b, Lorentz time component, write (N,257)
// ---------------------------------------------------------------------------
__global__ __launch_bounds__(256)
void final_kernel(const __bf16* __restrict__ phi_q, const __bf16* __restrict__ vs_bf,
                  const float* __restrict__ ktv, const float* __restrict__ ksum,
                  const float* __restrict__ v_map_w, const float* __restrict__ v_map_b,
                  const float* __restrict__ final_w, const float* __restrict__ final_b,
                  float* __restrict__ out) {
  const int n0 = blockIdx.x * 32;
  const int tid = threadIdx.x;
  const int wave = tid >> 5, lane = tid & 31;
  const int rowtile = wave & 1;
  const int colbase = (wave >> 1) * 64;
  const int crow0 = (lane >> 4) * 8, ccol = lane & 15;

  __shared__ __attribute__((aligned(32))) __bf16 At[32 * 256];  // [row][k] A operand
  __shared__ __attribute__((aligned(32))) __bf16 Bt[256 * 32];  // [col][k] B tile
  __shared__ float ksLds[256];
  __shared__ float denom[32];
  __shared__ float s2[32];

  v8f outacc[4];
#pragma unroll
  for (int t = 0; t < 4; ++t)
#pragma unroll
    for (int i = 0; i < 8; ++i) outacc[t][i] = 0.0f;

  for (int h = 0; h < NH; ++h) {
    __syncthreads();
    for (int i = tid; i < (32 * 256) / 8; i += 256) {   // v8bf copy phi_q -> At
      int m8 = (i & 31) * 8, r = i >> 5;
      *(v8bf*)&At[r * 256 + m8] =
          *(const v8bf*)&phi_q[(size_t)(n0 + r) * HD + h * DOUT + m8];
    }
    ksLds[tid] = ksum[h * DOUT + tid];
    if (tid < 32) denom[tid] = 0.0f;
    __syncthreads();

    { // denominator: 8 threads per row, 32 cols each
      int r = tid >> 3, seg = tid & 7;
      float p = 0.0f;
#pragma unroll
      for (int j = 0; j < 32; ++j) {
        int m = seg * 32 + j;
        p += (float)At[r * 256 + m] * ksLds[m];
      }
      atomicAdd(&denom[r], p);
    }

    v8f acc[4];
#pragma unroll
    for (int t = 0; t < 4; ++t)
#pragma unroll
      for (int i = 0; i < 8; ++i) acc[t][i] = 0.0f;

    // GEMM1: num = phi_q (32x256) @ ktv[h] (256x256)
    for (int kt = 0; kt < 8; ++kt) {
      __syncthreads();
      for (int i = tid; i < (256 * 32) / 4; i += 256) { // v4f loads of ktv rows
        int d4 = (i & 63) * 4, row = i >> 6;
        v4f v = *(const v4f*)&ktv[((size_t)h * DOUT + kt * 32 + row) * DOUT + d4];
#pragma unroll
        for (int j = 0; j < 4; ++j) Bt[(d4 + j) * 32 + row] = (__bf16)v[j];
      }
      __syncthreads();
      v16bf a = load_frag_a(At + rowtile * 16 * 256 + kt * 32, 256);
#pragma unroll
      for (int t = 0; t < 4; ++t) {
        v16bf b = load_frag_b32(Bt + (colbase + t * 16) * 32);
        acc[t] = WMMA_BF16(a, b, acc[t]);
      }
    }
    __syncthreads();

    // divide by denominator (one reciprocal per row)
#pragma unroll
    for (int r = 0; r < 8; ++r) {
      int row = rowtile * 16 + crow0 + r;
      float invd = 1.0f / (denom[row] + EPSF);
#pragma unroll
      for (int t = 0; t < 4; ++t) acc[t][r] *= invd;
    }

    // GEMM2: attn += vs (32x256) @ v_map_w^T (C-input fusion)
    for (int i = tid; i < (32 * 256) / 8; i += 256) {   // v8bf copy vs -> At
      int m8 = (i & 31) * 8, r = i >> 5;
      *(v8bf*)&At[r * 256 + m8] =
          *(const v8bf*)&vs_bf[(size_t)(n0 + r) * HD + h * DOUT + m8];
    }
    for (int kt = 0; kt < 8; ++kt) {
      __syncthreads();
      for (int i = tid; i < (256 * 32) / 4; i += 256) { // Bt[d][kk] = v_map_w[d, kt*32+kk]
        int kk4 = (i & 7) * 4, d = i >> 3;
        v4f v = *(const v4f*)&v_map_w[(size_t)d * 256 + kt * 32 + kk4];
        v4bf w;
#pragma unroll
        for (int j = 0; j < 4; ++j) w[j] = (__bf16)v[j];
        *(v4bf*)&Bt[d * 32 + kk4] = w;
      }
      __syncthreads();
      v16bf a = load_frag_a(At + rowtile * 16 * 256 + kt * 32, 256);
#pragma unroll
      for (int t = 0; t < 4; ++t) {
        v16bf b = load_frag_b32(Bt + (colbase + t * 16) * 32);
        acc[t] = WMMA_BF16(a, b, acc[t]);
      }
    }
    __syncthreads();

    // + v_map_b, stage attn as bf16 A operand
    float vb[4];
#pragma unroll
    for (int t = 0; t < 4; ++t) vb[t] = v_map_b[colbase + t * 16 + ccol];
#pragma unroll
    for (int t = 0; t < 4; ++t) {
      int col = colbase + t * 16 + ccol;
#pragma unroll
      for (int r = 0; r < 8; ++r) {
        int row = rowtile * 16 + crow0 + r;
        At[row * 256 + col] = (__bf16)(acc[t][r] + vb[t]);
      }
    }
    __syncthreads();

    // GEMM3: out += attn (32x256) @ final_w[:, h*256:(h+1)*256]^T
    for (int kt = 0; kt < 8; ++kt) {
      __syncthreads();
      for (int i = tid; i < (256 * 32) / 4; i += 256) { // Bt[o][kk] = final_w[o, h*256+kt*32+kk]
        int kk4 = (i & 7) * 4, o = i >> 3;
        v4f v = *(const v4f*)&final_w[(size_t)o * HD + h * DOUT + kt * 32 + kk4];
        v4bf w;
#pragma unroll
        for (int j = 0; j < 4; ++j) w[j] = (__bf16)v[j];
        *(v4bf*)&Bt[o * 32 + kk4] = w;
      }
      __syncthreads();
      v16bf a = load_frag_a(At + rowtile * 16 * 256 + kt * 32, 256);
#pragma unroll
      for (int t = 0; t < 4; ++t) {
        v16bf b = load_frag_b32(Bt + (colbase + t * 16) * 32);
        outacc[t] = WMMA_BF16(a, b, outacc[t]);
      }
    }
    __syncthreads();
  }

  // epilogue: bias, Lorentz time component, write (N, 257)
  if (tid < 32) s2[tid] = 0.0f;
  __syncthreads();
  float fb[4];
#pragma unroll
  for (int t = 0; t < 4; ++t) fb[t] = final_b[colbase + t * 16 + ccol];
#pragma unroll
  for (int r = 0; r < 8; ++r) {
    int row = rowtile * 16 + crow0 + r;
    float ss = 0.0f;
#pragma unroll
    for (int t = 0; t < 4; ++t) {
      float v = outacc[t][r] + fb[t];
      outacc[t][r] = v;
      ss += v * v;
    }
    atomicAdd(&s2[row], ss);
  }
  __syncthreads();
#pragma unroll
  for (int r = 0; r < 8; ++r) {
    int row = rowtile * 16 + crow0 + r;
#pragma unroll
    for (int t = 0; t < 4; ++t) {
      int col = colbase + t * 16 + ccol;
      out[(size_t)(n0 + row) * (DINP) + 1 + col] = outacc[t][r];
    }
  }
  if (tid < 32) out[(size_t)(n0 + tid) * (DINP)] = sqrtf(s2[tid] + KCURV);
}

// ---------------------------------------------------------------------------
extern "C" void kernel_launch(void* const* d_in, const int* in_sizes, int n_in,
                              void* d_out, int out_size, void* d_ws, size_t ws_size,
                              hipStream_t stream) {
  const float* query   = (const float*)d_in[0];
  const float* source  = (const float*)d_in[1];
  const float* Wq_w    = (const float*)d_in[2];
  const float* Wq_b    = (const float*)d_in[3];
  const float* Wk_w    = (const float*)d_in[4];
  const float* Wk_b    = (const float*)d_in[5];
  const float* Wv_w    = (const float*)d_in[6];
  const float* Wv_b    = (const float*)d_in[7];
  const float* nscale  = (const float*)d_in[8];
  const float* v_map_w = (const float*)d_in[9];
  const float* v_map_b = (const float*)d_in[10];
  const float* final_w = (const float*)d_in[11];
  const float* final_b = (const float*)d_in[12];
  float* out = (float*)d_out;

  char* ws = (char*)d_ws;
  size_t off = 0;
  __bf16* phi_q = (__bf16*)(ws + off); off += (size_t)NTOK * HD * 2;
  __bf16* phi_k = (__bf16*)(ws + off); off += (size_t)NTOK * HD * 2;
  __bf16* vs_bf = (__bf16*)(ws + off); off += (size_t)NTOK * HD * 2;
  float*  ktv   = (float*)(ws + off);  off += (size_t)NH * DOUT * DOUT * 4;
  float*  ksum  = (float*)(ws + off);  off += (size_t)NH * DOUT * 4;

  hipMemsetAsync(ktv, 0, ((size_t)NH * DOUT * DOUT + NH * DOUT) * 4, stream);

  proj_phi_q_kernel<<<dim3(NTOK / 32, NH), 256, 0, stream>>>(query, Wq_w, Wq_b, nscale, phi_q);
  proj_kv_kernel<<<dim3(NTOK / 32, NH), 256, 0, stream>>>(source, Wk_w, Wk_b, Wv_w, Wv_b,
                                                          nscale, phi_k, vs_bf);
  ktv_kernel<<<dim3(NH, 4, SPLITK), 256, 0, stream>>>(phi_k, vs_bf, ktv);
  ksum_kernel<<<dim3(NH, KSPLIT), 256, 0, stream>>>(phi_k, ksum);
  final_kernel<<<dim3(NTOK / 32), 256, 0, stream>>>(phi_q, vs_bf, ktv, ksum,
                                                    v_map_w, v_map_b, final_w, final_b, out);
}
